// SelfAttention_5652176961587
// MI455X (gfx1250) — matrix operand order repaired and checked by reading, not compiled, not run on hardware
//
#include <hip/hip_runtime.h>
#include <hip/hip_bf16.h>

typedef __attribute__((ext_vector_type(16))) _Float16 v16h;
typedef __attribute__((ext_vector_type(8)))  _Float16 v8h;
typedef __attribute__((ext_vector_type(8)))  float    v8f;

#define NUM_HEADS 16
#define KEY_SIZE  64
#define D_MODEL   1024
#define QKV_W     3072      // 16 * 192
#define BATCH     2
#define SEQ       2048
#define MTOT      (BATCH * SEQ)   // 4096

// ---- CDNA5 async global->LDS path (guarded; fallback = sync copy) ----------
#if defined(__has_builtin)
#if __has_builtin(__builtin_amdgcn_global_load_async_to_lds_b128) && \
    __has_builtin(__builtin_amdgcn_s_wait_asynccnt)
#define HAS_ASYNC 1
#endif
#endif

typedef __attribute__((vector_size(4 * sizeof(int)))) int v4i_vs;
typedef __attribute__((address_space(1))) v4i_vs* gv4i_ptr;   // global (AS1)
typedef __attribute__((address_space(3))) v4i_vs* lv4i_ptr;   // LDS (AS3)

static __device__ __forceinline__ void copy16(const _Float16* g, _Float16* l) {
#ifdef HAS_ASYNC
    __builtin_amdgcn_global_load_async_to_lds_b128((gv4i_ptr)g, (lv4i_ptr)l, 0, 0);
#else
    *(v8h*)l = *(const v8h*)g;
#endif
}
static __device__ __forceinline__ void wait_async_le8() {
#ifdef HAS_ASYNC
    __builtin_amdgcn_s_wait_asynccnt(8);
#endif
}
static __device__ __forceinline__ void wait_async_0() {
#ifdef HAS_ASYNC
    __builtin_amdgcn_s_wait_asynccnt(0);
#endif
}

static __device__ __forceinline__ v16h combine16(v8h lo, v8h hi) {
    v16h r;
#pragma unroll
    for (int j = 0; j < 8; ++j) { r[j] = lo[j]; r[j + 8] = hi[j]; }
    return r;
}

static __device__ __forceinline__ v8f wmma_f16(v16h a, v16h b, v8f c) {
    return __builtin_amdgcn_wmma_f32_16x16x32_f16(false, a, false, b,
                                                  (short)0, c, false, false);
}

// ---------------------------------------------------------------------------
// Kernel 0a: X f32 -> Xh f16 (same layout). Memory bound, vectorized.
// ---------------------------------------------------------------------------
__global__ __launch_bounds__(256) void cvt_x_kernel(
    const float* __restrict__ X, _Float16* __restrict__ Xh)
{
    const size_t i = ((size_t)blockIdx.x * 256 + threadIdx.x) * 8;
    const float* s = X + i;
    v8h h;
#pragma unroll
    for (int j = 0; j < 8; ++j) h[j] = (_Float16)s[j];
    *(v8h*)(Xh + i) = h;
}

// ---------------------------------------------------------------------------
// Kernel 0b: W f32 [1024][3072] -> Wt f16 [3072][1024] (transposed via LDS).
// Makes GEMM B-tiles row-contiguous so they can be async-copied to LDS.
// ---------------------------------------------------------------------------
__global__ __launch_bounds__(256) void cvt_wt_kernel(
    const float* __restrict__ W, _Float16* __restrict__ Wt)
{
    __shared__ float tile[32][33];
    const int tx = threadIdx.x & 31;
    const int ty = threadIdx.x >> 5;       // 0..7
    const int n0 = blockIdx.x * 32;
    const int k0 = blockIdx.y * 32;
#pragma unroll
    for (int j = 0; j < 4; ++j)
        tile[ty + 8 * j][tx] = W[(size_t)(k0 + ty + 8 * j) * QKV_W + n0 + tx];
    __syncthreads();
#pragma unroll
    for (int j = 0; j < 4; ++j)
        Wt[(size_t)(n0 + ty + 8 * j) * D_MODEL + k0 + tx] =
            (_Float16)tile[tx][ty + 8 * j];
}

// ---------------------------------------------------------------------------
// Kernel 1: qkv = f16( Xh @ Wt^T + bias ), stored [4096][3072] f16.
// Block tile 128x128, 8 waves (4x2), wave tile 32x64. K-step 64 (2 WMMA
// K-chunks per round, 16 WMMA per barrier pair). Double-buffered LDS with
// async global->LDS staging pipelined one K-round ahead (s_wait_asynccnt).
// LDS rows stride 72 halfs (144B = 9*16B: b128-aligned, banks spread).
// ---------------------------------------------------------------------------
__global__ __launch_bounds__(256) void qkv_gemm_kernel(
    const _Float16* __restrict__ Xh, const _Float16* __restrict__ Wt,
    const float* __restrict__ bias, _Float16* __restrict__ qkv)
{
    __shared__ _Float16 ldsA[2][128 * 72];
    __shared__ _Float16 ldsB[2][128 * 72];

    const int t    = threadIdx.x;
    const int lane = t & 31;
    const int wid  = t >> 5;
    const int hl   = lane >> 4;
    const int ln   = lane & 15;
    const int m0   = blockIdx.x * 128;
    const int n0   = blockIdx.y * 128;
    const int wm   = wid >> 1;        // 0..3
    const int wn   = wid & 1;         // 0..1

    // staging assignment: thread covers 64B (4 x b128) of one row of A and B
    const int srow = t >> 1;                   // 0..127
    const int sseg = (t & 1) * 4;              // segments sseg..sseg+3 (8 halfs each)
    const _Float16* gA = Xh + (size_t)(m0 + srow) * D_MODEL + sseg * 8;
    const _Float16* gB = Wt + (size_t)(n0 + srow) * D_MODEL + sseg * 8;

    auto stage = [&](int buf, int k0) {
#pragma unroll
        for (int i = 0; i < 4; ++i) {
            copy16(gA + k0 + i * 8, &ldsA[buf][srow * 72 + (sseg + i) * 8]);
            copy16(gB + k0 + i * 8, &ldsB[buf][srow * 72 + (sseg + i) * 8]);
        }
    };

    v8f acc[2][4] = {};

    stage(0, 0);
    for (int r = 0; r < D_MODEL / 64; ++r) {
        const int buf = r & 1;
        if (r + 1 < D_MODEL / 64) {
            stage(buf ^ 1, (r + 1) * 64);   // prefetch next round
            wait_async_le8();               // round r's 8 copies landed
        } else {
            wait_async_0();
        }
        __syncthreads();

        v16h afr[2][2], bfr[2][4];
#pragma unroll
        for (int ks = 0; ks < 2; ++ks) {
#pragma unroll
            for (int mt = 0; mt < 2; ++mt) {
                const _Float16* p = &ldsA[buf][(wm * 32 + mt * 16 + ln) * 72 + ks * 32];
                afr[ks][mt] = combine16(*(v8h*)(p + hl * 8), *(v8h*)(p + 16 + hl * 8));
            }
#pragma unroll
            for (int nt = 0; nt < 4; ++nt) {
                const _Float16* p = &ldsB[buf][(wn * 64 + nt * 16 + ln) * 72 + ks * 32];
                bfr[ks][nt] = combine16(*(v8h*)(p + hl * 8), *(v8h*)(p + 16 + hl * 8));
            }
        }
#pragma unroll
        for (int ks = 0; ks < 2; ++ks)
#pragma unroll
            for (int mt = 0; mt < 2; ++mt)
#pragma unroll
                for (int nt = 0; nt < 4; ++nt)
                    acc[mt][nt] = wmma_f16(afr[ks][mt], bfr[ks][nt], acc[mt][nt]);
        __syncthreads();
    }

    // epilogue: +bias, convert to f16
#pragma unroll
    for (int mt = 0; mt < 2; ++mt)
#pragma unroll
        for (int nt = 0; nt < 4; ++nt) {
            const int n = n0 + wn * 64 + nt * 16 + ln;
            const float bb = bias[n];
#pragma unroll
            for (int i = 0; i < 8; ++i) {
                const int m = m0 + wm * 32 + mt * 16 + i + 8 * hl;
                qkv[(size_t)m * QKV_W + n] = (_Float16)(acc[mt][nt][i] + bb);
            }
        }
}

// ---------------------------------------------------------------------------
// Kernel 2: flash attention. Block = (b, h, 128-query tile); 8 waves x 16 rows.
// Key chunks of 32: S = Q*K^T (4 WMMA), online softmax, O += P*V (4 WMMA).
// K chunk staged via async global->LDS; V staged transposed manually.
// Reference multiplies scores by sqrt(64) = 8.
// ---------------------------------------------------------------------------
__global__ __launch_bounds__(256) void attn_kernel(
    const _Float16* __restrict__ qkv, float* __restrict__ out)
{
    __shared__ _Float16 ldsK[32 * 64];        // [key][feat]
    __shared__ _Float16 ldsV[64 * 32];        // [feat][key] (transposed)
    __shared__ _Float16 ldsP[8 * 16 * 40];    // per-wave P bounce, stride 40

    const int t    = threadIdx.x;
    const int lane = t & 31;
    const int wid  = t >> 5;
    const int hl   = lane >> 4;
    const int ln   = lane & 15;
    const int qt   = blockIdx.x;   // 0..15
    const int h    = blockIdx.y;   // 0..15
    const int b    = blockIdx.z;   // 0..1
    const int q0   = qt * 128 + wid * 16;

    // Q fragments (16 rows x 64 feat, two 32-wide K-steps) straight from global
    const int qrow = q0 + ln;
    const _Float16* qptr = qkv + (size_t)(b * SEQ + qrow) * QKV_W + h * 192;
    v16h qa[2];
#pragma unroll
    for (int ks = 0; ks < 2; ++ks) {
        const int kf0 = ks * 32;
        qa[ks] = combine16(*(const v8h*)(qptr + kf0 + hl * 8),
                           *(const v8h*)(qptr + kf0 + 16 + hl * 8));
    }

    float mrow[8], lrow[8];
    v8f o[4] = {};
#pragma unroll
    for (int i = 0; i < 8; ++i) { mrow[i] = -1e30f; lrow[i] = 0.0f; }

    _Float16* pp = &ldsP[wid * 16 * 40];

    for (int kc = 0; kc < SEQ; kc += 32) {
        __syncthreads();   // previous round's readers done with ldsK/ldsV
        {
            const int key  = t >> 3;
            const int part = t & 7;
            const _Float16* kvp = qkv + (size_t)(b * SEQ + kc + key) * QKV_W + h * 192;
            copy16(kvp + 64 + part * 8, &ldsK[key * 64 + part * 8]);   // K: async
            v8h vv = *(const v8h*)(kvp + 128 + part * 8);              // V: transpose
#pragma unroll
            for (int j = 0; j < 8; ++j)
                ldsV[(part * 8 + j) * 32 + key] = vv[j];
        }
        wait_async_0();
        __syncthreads();

        // scores S = Q * K^T for 32 keys (2 n-tiles x 2 k-steps)
        v8f s[2] = {};
#pragma unroll
        for (int nt = 0; nt < 2; ++nt)
#pragma unroll
            for (int ks = 0; ks < 2; ++ks) {
                const _Float16* kb = &ldsK[(nt * 16 + ln) * 64 + ks * 32];
                v16h bf = combine16(*(const v8h*)(kb + hl * 8), *(const v8h*)(kb + 16 + hl * 8));
                s[nt] = wmma_f16(qa[ks], bf, s[nt]);
            }

        // online softmax (row M = i + 8*hl lives across the 16 lanes of a half)
#pragma unroll
        for (int i = 0; i < 8; ++i) {
            float x0 = s[0][i] * 8.0f;   // * sqrt(KEY_SIZE)
            float x1 = s[1][i] * 8.0f;
            float cm = fmaxf(x0, x1);
            cm = fmaxf(cm, __shfl_xor(cm, 1, 32));
            cm = fmaxf(cm, __shfl_xor(cm, 2, 32));
            cm = fmaxf(cm, __shfl_xor(cm, 4, 32));
            cm = fmaxf(cm, __shfl_xor(cm, 8, 32));
            const float mn = fmaxf(mrow[i], cm);
            const float sc = __expf(mrow[i] - mn);
            const float p0 = __expf(x0 - mn);
            const float p1 = __expf(x1 - mn);
            float rs = p0 + p1;
            rs += __shfl_xor(rs, 1, 32);
            rs += __shfl_xor(rs, 2, 32);
            rs += __shfl_xor(rs, 4, 32);
            rs += __shfl_xor(rs, 8, 32);
            lrow[i] = lrow[i] * sc + rs;
            mrow[i] = mn;
#pragma unroll
            for (int vt = 0; vt < 4; ++vt) o[vt][i] *= sc;
            pp[(i + 8 * hl) * 40 + ln]      = (_Float16)p0;
            pp[(i + 8 * hl) * 40 + 16 + ln] = (_Float16)p1;
        }
        asm volatile("s_wait_dscnt 0" ::: "memory");  // intra-wave P store->load

        // P (16x32) as A-fragment, V tiles as B-fragments: O += P*V
        v16h pa = combine16(*(v8h*)&pp[ln * 40 + hl * 8],
                            *(v8h*)&pp[ln * 40 + 16 + hl * 8]);
#pragma unroll
        for (int vt = 0; vt < 4; ++vt) {
            const _Float16* vb = &ldsV[(vt * 16 + ln) * 32];
            v16h bf = combine16(*(const v8h*)(vb + hl * 8), *(const v8h*)(vb + 16 + hl * 8));
            o[vt] = wmma_f16(pa, bf, o[vt]);
        }
    }

    // normalize and write out[b][q][h*64 + feat] (f32)
#pragma unroll
    for (int i = 0; i < 8; ++i) {
        const int   qg  = q0 + i + 8 * hl;
        const float inv = 1.0f / lrow[i];
        float* op = out + (size_t)(b * SEQ + qg) * (NUM_HEADS * KEY_SIZE) + h * 64;
#pragma unroll
        for (int vt = 0; vt < 4; ++vt)
            op[vt * 16 + ln] = o[vt][i] * inv;
    }
}

extern "C" void kernel_launch(void* const* d_in, const int* in_sizes, int n_in,
                              void* d_out, int out_size, void* d_ws, size_t ws_size,
                              hipStream_t stream) {
    const float* X    = (const float*)d_in[0];   // (2,2048,1024) f32
    const float* W    = (const float*)d_in[1];   // (1024,3072) f32
    const float* bias = (const float*)d_in[2];   // (3072,) f32
    float* out = (float*)d_out;                  // (2,2048,1024) f32

    // workspace layout (f16): qkv 24MB | Xh 8MB | Wt 6MB  (total ~38MB)
    char* ws = (char*)d_ws;
    _Float16* qkv = (_Float16*)ws;
    _Float16* Xh  = (_Float16*)(ws + (size_t)MTOT * QKV_W * 2);
    _Float16* Wt  = (_Float16*)(ws + (size_t)MTOT * QKV_W * 2
                                   + (size_t)MTOT * D_MODEL * 2);

    cvt_x_kernel<<<dim3((MTOT * D_MODEL) / (8 * 256)), 256, 0, stream>>>(X, Xh);
    cvt_wt_kernel<<<dim3(QKV_W / 32, D_MODEL / 32), 256, 0, stream>>>(W, Wt);
    qkv_gemm_kernel<<<dim3(MTOT / 128, QKV_W / 128), 256, 0, stream>>>(Xh, Wt, bias, qkv);
    attn_kernel<<<dim3(SEQ / 128, NUM_HEADS, BATCH), 256, 0, stream>>>(qkv, out);
}
